// ReadoutLayer1_60696477827519
// MI455X (gfx1250) — matrix operand order repaired and checked by reading, not compiled, not run on hardware
//
#include <hip/hip_runtime.h>
#include <cstdint>
#include <math.h>

// Problem dims (fixed by reference)
#define BB 8
#define SS 512
#define NN 128
#define DD 256      // D_IN (= K of main GEMM)
#define HH 128      // D_HID (= N of main GEMM)
#define DOUT 64
#define ROWS (BB*SS*NN)          // 524288 GEMM rows
#define TILES (ROWS/16)          // 32768 16-row tiles

typedef __attribute__((ext_vector_type(4)))  float    v4f;
typedef __attribute__((ext_vector_type(8)))  float    v8f;
typedef __attribute__((ext_vector_type(16))) _Float16 v16h;

// ---------------------------------------------------------------------------
// Kernel 1: scores[row] = mask(b,s) * sum_h( w_len[h] * relu(x_row . W1[h,:] + b1[h]) )
// f16 WMMA (V_WMMA_F32_16X16X32_F16), fp32 accumulate.  K-loop: 8 steps of 32,
// unrolled x2 only (full unroll made all 64 B-fragments live -> scratch spills).
//
// A (16x32 f16), contiguous K order (same as B): lane m=lane&15 -> row M.
//   lane<16:  VGPR0-7 = K 0..15
//   lane>=16: VGPR0-7 = K 16..31
//   -> per lane: one contiguous 16-float chunk of its x row at
//      k0 + 16*grp, converted fp32 -> f16 in-register.
// B (32x16 f16): lane m -> col N; lane<16 holds K 0..15, lane>=16 K 16..31
//   -> one contiguous 16-half (32B) read from W1 staged as f16 in LDS,
//      natural [n][k] layout.
// C (16x16 f32): VGPR r, lane -> M = r + 8*grp, N = (lane&15) + 16*nt.
// ---------------------------------------------------------------------------
__global__ __launch_bounds__(256) void scores_kernel(
    const float* __restrict__ x, const int* __restrict__ seq_len,
    const float* __restrict__ W1, const float* __restrict__ b1,
    const float* __restrict__ wlen, float* __restrict__ scores)
{
  extern __shared__ char smem_raw[];
  _Float16* lw1 = (_Float16*)smem_raw;                    // HH*DD halves (64KB)
  float*    lb1 = (float*)(smem_raw + (size_t)HH*DD*2);   // HH floats
  float*    lwl = lb1 + HH;                               // HH floats

  const int tid = threadIdx.x;
  for (int i = tid; i < HH * DD; i += 256) lw1[i] = (_Float16)W1[i];
  for (int i = tid; i < HH; i += 256) { lb1[i] = b1[i]; lwl[i] = wlen[i]; }
  __syncthreads();

  const int wave = tid >> 5;
  const int lane = tid & 31;
  const int m    = lane & 15;         // M row (A/C) or N col (B)
  const int grp  = lane >> 4;         // half-wave group

  for (int tile = blockIdx.x * 8 + wave; tile < TILES; tile += gridDim.x * 8) {
    // Block LICM from hoisting the (tile-invariant) LDS B-fragment loads out
    // of this loop -- that hoist is what caused 512-VGPR spills to scratch.
    asm volatile("" ::: "memory");

    const int row0 = tile << 4;
    // per-lane base into this lane's x row, at its K-group offset
    const float* xr = x + (size_t)(row0 + m) * DD + 16 * grp;

    v8f c[8];
    #pragma unroll
    for (int nt = 0; nt < 8; ++nt) {
      v8f z = {0.f,0.f,0.f,0.f,0.f,0.f,0.f,0.f};
      c[nt] = z;
    }

    #pragma unroll 2
    for (int k0 = 0; k0 < DD; k0 += 32) {
      // ---- A fragment: 16 fp32 -> v16h ----
      v4f a0 = *(const v4f*)(xr + k0);          // K: k0 + 16g + 0..3
      v4f a1 = *(const v4f*)(xr + k0 + 4);      // K: k0 + 16g + 4..7
      v4f a2 = *(const v4f*)(xr + k0 + 8);      // K: k0 + 16g + 8..11
      v4f a3 = *(const v4f*)(xr + k0 + 12);     // K: k0 + 16g + 12..15
      v16h a;
      #pragma unroll
      for (int i = 0; i < 4; ++i) {
        a[i]      = (_Float16)a0[i];
        a[4 + i]  = (_Float16)a1[i];
        a[8 + i]  = (_Float16)a2[i];
        a[12 + i] = (_Float16)a3[i];
      }
      // ---- 8 B fragments from LDS, one WMMA each ----
      const _Float16* wb = lw1 + k0 + 16 * grp;
      #pragma unroll
      for (int nt = 0; nt < 8; ++nt) {
        v16h bfrag = *(const v16h*)(wb + (size_t)(nt * 16 + m) * DD);
        c[nt] = __builtin_amdgcn_wmma_f32_16x16x32_f16(
            false, a, false, bfrag, (short)0, c[nt], false, false);
      }
    }

    // Epilogue: bias + relu + dot with w_len over all 128 hidden units.
    float part[8];
    #pragma unroll
    for (int r = 0; r < 8; ++r) part[r] = 0.f;
    #pragma unroll
    for (int nt = 0; nt < 8; ++nt) {
      const int n = nt * 16 + m;
      const float bias = lb1[n];
      const float wl   = lwl[n];
      #pragma unroll
      for (int r = 0; r < 8; ++r) {
        float v = c[nt][r] + bias;
        v = v > 0.f ? v : 0.f;
        part[r] += v * wl;
      }
    }
    // Sum across the 16 lanes of each half-group (xor 1,2,4,8 stays in-group).
    #pragma unroll
    for (int r = 0; r < 8; ++r) {
      part[r] += __shfl_xor(part[r], 1, 32);
      part[r] += __shfl_xor(part[r], 2, 32);
      part[r] += __shfl_xor(part[r], 4, 32);
      part[r] += __shfl_xor(part[r], 8, 32);
    }

    // Mask: all 16 rows of a tile share (b, s) since 128 % 16 == 0.
    const int s = (row0 >> 7) & (SS - 1);
    const int b = row0 >> 16;               // / (SS*NN)
    const float mask = (s < seq_len[b]) ? 1.0f : -10.0f;

    if (m == 0) {
      #pragma unroll
      for (int r = 0; r < 8; ++r)
        scores[row0 + grp * 8 + r] = part[r] * mask;
    }
  }
}

// ---------------------------------------------------------------------------
// Kernel 2: per (b, node): softmax over s of scores, then
//           pooled[b,n,d] = sum_s attn[s] * x[b,s,n,d].  Second pass over x,
//           done in full fp32 from the original input.
// ---------------------------------------------------------------------------
__global__ __launch_bounds__(256) void softmax_pool_kernel(
    const float* __restrict__ x, const float* __restrict__ scores,
    float* __restrict__ pooled)
{
  __shared__ float ebuf[SS];
  __shared__ float red[256];
  const int b = blockIdx.x >> 7;
  const int n = blockIdx.x & (NN - 1);
  const int tid = threadIdx.x;

  const float s0 = scores[(size_t)(b * SS + tid)       * NN + n];
  const float s1 = scores[(size_t)(b * SS + tid + 256) * NN + n];

  red[tid] = fmaxf(s0, s1);
  __syncthreads();
  for (int off = 128; off > 0; off >>= 1) {
    if (tid < off) red[tid] = fmaxf(red[tid], red[tid + off]);
    __syncthreads();
  }
  const float mx = red[0];
  __syncthreads();

  const float e0 = expf(s0 - mx);
  const float e1 = expf(s1 - mx);
  ebuf[tid]       = e0;
  ebuf[tid + 256] = e1;
  red[tid] = e0 + e1;
  __syncthreads();
  for (int off = 128; off > 0; off >>= 1) {
    if (tid < off) red[tid] += red[tid + off];
    __syncthreads();
  }
  const float inv = 1.0f / red[0];

  // Each thread owns one d; coalesced 1KB rows across the block.
  const int d = tid;
  const float* xb = x + ((size_t)(b * SS) * NN + n) * DD + d;
  float acc = 0.f;
  #pragma unroll 4
  for (int s = 0; s < SS; ++s) {
    if (s + 8 < SS)
      __builtin_prefetch(xb + (size_t)(s + 8) * NN * DD, 0, 0);  // global_prefetch_b8
    acc += ebuf[s] * xb[(size_t)s * NN * DD];
  }
  pooled[(size_t)blockIdx.x * DD + d] = acc * inv;
}

// ---------------------------------------------------------------------------
// Kernel 3: gated readout per (b,n) row:
//   g = (sigmoid(p.Wa^T+ba) * tanh(p.We^T+be)) . Wf^T + bf   -> (1024, 64)
// ---------------------------------------------------------------------------
__global__ __launch_bounds__(256) void readout_kernel(
    const float* __restrict__ pooled,
    const float* __restrict__ Wa, const float* __restrict__ ba,
    const float* __restrict__ We, const float* __restrict__ be,
    const float* __restrict__ Wf, const float* __restrict__ bf,
    float* __restrict__ g)
{
  __shared__ float p[DD];
  __shared__ float t[DD];
  const int row = blockIdx.x;
  const int tid = threadIdx.x;

  p[tid] = pooled[(size_t)row * DD + tid];
  __syncthreads();

  float accA = ba[tid];
  float accE = be[tid];
  const float* wa = Wa + (size_t)tid * DD;
  const float* we = We + (size_t)tid * DD;
  #pragma unroll 4
  for (int k = 0; k < DD; ++k) {
    const float pk = p[k];
    accA += pk * wa[k];
    accE += pk * we[k];
  }
  const float att = 1.0f / (1.0f + expf(-accA));
  const float emb = tanhf(accE);
  t[tid] = att * emb;
  __syncthreads();

  if (tid < DOUT) {
    float acc = bf[tid];
    const float* wf = Wf + (size_t)tid * DD;
    #pragma unroll 4
    for (int k = 0; k < DD; ++k) acc += t[k] * wf[k];
    g[(size_t)row * DOUT + tid] = acc;
  }
}

// ---------------------------------------------------------------------------
// Kernel 4: out[b,d] = sum_n g[b,n,d] / N + max_n g[b,n,d]   (deterministic)
// ---------------------------------------------------------------------------
__global__ void finalize_kernel(const float* __restrict__ g,
                                float* __restrict__ out)
{
  const int b = blockIdx.x;
  const int d = threadIdx.x;  // 64 threads
  float sum = 0.f;
  float mx = -INFINITY;
  for (int n = 0; n < NN; ++n) {
    const float v = g[((size_t)(b * NN + n)) * DOUT + d];
    sum += v;
    mx = fmaxf(mx, v);
  }
  out[b * DOUT + d] = sum / (float)NN + mx;
}

// ---------------------------------------------------------------------------
extern "C" void kernel_launch(void* const* d_in, const int* in_sizes, int n_in,
                              void* d_out, int out_size, void* d_ws, size_t ws_size,
                              hipStream_t stream) {
  const float* x    = (const float*)d_in[0];
  const int*   seq  = (const int*)  d_in[1];
  const float* W1   = (const float*)d_in[2];
  const float* b1   = (const float*)d_in[3];
  const float* wlen = (const float*)d_in[4];
  const float* Wa   = (const float*)d_in[5];
  const float* ba   = (const float*)d_in[6];
  const float* We   = (const float*)d_in[7];
  const float* be   = (const float*)d_in[8];
  const float* Wf   = (const float*)d_in[9];
  const float* bf   = (const float*)d_in[10];
  float* out = (float*)d_out;

  float* ws      = (float*)d_ws;
  float* scores  = ws;                          // ROWS floats        (2 MB)
  float* pooled  = scores + ROWS;               // BB*NN*DD floats    (1 MB)
  float* gbuf    = pooled + (size_t)BB*NN*DD;   // BB*NN*DOUT floats  (256 KB)

  const size_t smem = (size_t)HH * DD * 2 + (size_t)2 * HH * 4;  // ~65 KB
  scores_kernel<<<1024, 256, smem, stream>>>(x, seq, W1, b1, wlen, scores);
  softmax_pool_kernel<<<BB * NN, 256, 0, stream>>>(x, scores, pooled);
  readout_kernel<<<BB * NN, 256, 0, stream>>>(pooled, Wa, ba, We, be, Wf, bf, gbuf);
  finalize_kernel<<<BB, DOUT, 0, stream>>>(gbuf, out);
}
